// TransformerBlock_35553739276450
// MI455X (gfx1250) — matrix operand order repaired and checked by reading, not compiled, not run on hardware
//
#include <hip/hip_runtime.h>
#include <hip/hip_bf16.h>

typedef __bf16 bf16;
typedef __attribute__((ext_vector_type(8)))  __bf16 v8bf;
typedef __attribute__((ext_vector_type(16))) __bf16 v16bf;
typedef __attribute__((ext_vector_type(8)))  float  v8f;

#define DIM     512
#define HEADS   8
#define HD      64
#define HIDDEN  2048
#define NTOK    16384     // B*P*N = 16*4*256
#define NSEQ    256
#define BP      64        // B*P
#define BPH     512       // B*P*HEADS

static __device__ inline v8f zero8() {
    v8f z = {0.f,0.f,0.f,0.f,0.f,0.f,0.f,0.f};
    return z;
}

// Two contiguous 16-byte loads -> one 16x32 bf16 operand fragment
static __device__ inline v16bf ld_frag(const bf16* p0, const bf16* p1) {
    v8bf lo = *(const v8bf*)p0;
    v8bf hi = *(const v8bf*)p1;
    return __builtin_shufflevector(lo, hi, 0,1,2,3,4,5,6,7,8,9,10,11,12,13,14,15);
}

static __device__ inline v8f wmma_bf16(v16bf a, v16bf b, v8f c) {
    // (neg_a, A, neg_b, B, c_mod, C, reuse_a, reuse_b)
    return __builtin_amdgcn_wmma_f32_16x16x32_bf16(false, a, false, b, (short)0, c, false, false);
}

static __device__ inline float gelu_exact(float x) {
    return 0.5f * x * (1.0f + erff(x * 0.70710678118654752f));
}

// ---------------- f32 -> bf16 convert ----------------
__global__ __launch_bounds__(256) void cvt_bf16(const float* __restrict__ in,
                                                bf16* __restrict__ out, int n) {
    int i = blockIdx.x * 256 + threadIdx.x;
    if (i < n) out[i] = (bf16)in[i];
}

// ---------------- LayerNorm: one wave32 per token ----------------
__global__ __launch_bounds__(256) void ln_kernel(const float* __restrict__ x,
                                                 const float* __restrict__ g,
                                                 const float* __restrict__ b,
                                                 bf16* __restrict__ out) {
    int wave = threadIdx.x >> 5, lane = threadIdx.x & 31;
    size_t token = (size_t)blockIdx.x * 8 + wave;
    const float* xp = x + token * DIM;
    float vals[16];
    float s = 0.f, s2 = 0.f;
#pragma unroll
    for (int i = 0; i < 16; i++) {
        float v = xp[lane + 32 * i];
        vals[i] = v; s += v; s2 += v * v;
    }
#pragma unroll
    for (int m = 16; m >= 1; m >>= 1) {
        s  += __shfl_xor(s,  m, 32);
        s2 += __shfl_xor(s2, m, 32);
    }
    float mu   = s * (1.f / DIM);
    float var  = s2 * (1.f / DIM) - mu * mu;
    float rstd = rsqrtf(var + 1e-5f);
    bf16* op = out + token * DIM;
#pragma unroll
    for (int i = 0; i < 16; i++) {
        int c = lane + 32 * i;
        op[c] = (bf16)((vals[i] - mu) * rstd * g[c] + b[c]);
    }
}

// ---------------- repack qkv -> Q(scaled), K, V^T ----------------
__global__ __launch_bounds__(256) void repack_qkv(const bf16* __restrict__ qkv,
                                                  bf16* __restrict__ qb,
                                                  bf16* __restrict__ kb,
                                                  bf16* __restrict__ vt) {
    int idx = blockIdx.x * 256 + threadIdx.x;       // over NTOK*DIM
    if (idx >= NTOK * DIM) return;
    int c = idx & (DIM - 1); int token = idx >> 9;
    int h = c >> 6;          int e = c & 63;
    int bp = token >> 8;     int n = token & 255;
    int bph = bp * HEADS + h;
    const bf16* src = qkv + (size_t)token * (3 * DIM);
    float qv = (float)src[c] * 0.125f;              // HD^-0.5 = 1/8
    qb[((size_t)bph * NSEQ + n) * HD + e] = (bf16)qv;
    kb[((size_t)bph * NSEQ + n) * HD + e] = src[DIM + c];
    vt[((size_t)bph * HD + e) * NSEQ + n] = src[2 * DIM + c];
}

// ---------------- generic GEMM: out = A(MxK) * B(NxK)^T ----------------
// Wave tile 64x64 (16 WMMA per 16 b128 loads), WG tile 128x256 (8 waves, 2x4).
// EPI 0: store bf16 | 1: +bias, GELU, store bf16 | 2: +bias +resid, store f32
template <int EPI>
__global__ __launch_bounds__(256) void gemm_kernel(const bf16* __restrict__ A,
                                                   const bf16* __restrict__ B,
                                                   int N, int K,
                                                   const float* __restrict__ bias,
                                                   const float* __restrict__ resid,
                                                   float* __restrict__ outF,
                                                   bf16* __restrict__ outB) {
    int tid = threadIdx.x, wave = tid >> 5, lane = tid & 31;
    int lm = lane & 15, kh = lane >> 4;
    int wm = wave >> 2, wn = wave & 3;
    size_t rowBase = (size_t)blockIdx.x * 128 + wm * 64;
    size_t colBase = (size_t)blockIdx.y * 256 + wn * 64;

    const bf16* ap[4];
    const bf16* bp[4];
#pragma unroll
    for (int i = 0; i < 4; i++) ap[i] = A + (rowBase + i * 16 + lm) * K;
#pragma unroll
    for (int j = 0; j < 4; j++) bp[j] = B + (colBase + j * 16 + lm) * K;

    v8f acc[4][4];
#pragma unroll
    for (int i = 0; i < 4; i++)
#pragma unroll
        for (int j = 0; j < 4; j++) acc[i][j] = zero8();

    for (int k = 0; k < K; k += 32) {
        v16bf a[4], b[4];
#pragma unroll
        for (int i = 0; i < 4; i++)
            a[i] = ld_frag(ap[i] + k + kh * 8, ap[i] + k + 16 + kh * 8);
#pragma unroll
        for (int j = 0; j < 4; j++)
            b[j] = ld_frag(bp[j] + k + kh * 8, bp[j] + k + 16 + kh * 8);
#pragma unroll
        for (int i = 0; i < 4; i++)
#pragma unroll
            for (int j = 0; j < 4; j++)
                acc[i][j] = wmma_bf16(a[i], b[j], acc[i][j]);
    }

#pragma unroll
    for (int i = 0; i < 4; i++)
#pragma unroll
        for (int j = 0; j < 4; j++)
#pragma unroll
            for (int v = 0; v < 8; v++) {
                size_t r = rowBase + i * 16 + v + 8 * kh;
                size_t c = colBase + j * 16 + lm;
                float val = acc[i][j][v];
                if (EPI == 0) {
                    outB[r * N + c] = (bf16)val;
                } else if (EPI == 1) {
                    val += bias[c];
                    outB[r * N + c] = (bf16)gelu_exact(val);
                } else {
                    val += bias[c] + resid[r * N + c];
                    outF[r * N + c] = val;
                }
            }
}

// ---------------- attention: one WG per (b,p,h, 64-query block) ----------------
__global__ __launch_bounds__(256) void attn_kernel(const bf16* __restrict__ qb,
                                                   const bf16* __restrict__ kb,
                                                   const bf16* __restrict__ vt,
                                                   bf16* __restrict__ obuf) {
    __shared__ bf16  Sb[64][NSEQ];      // 32 KB: scores, then exp(P) in-place
    __shared__ float red[64][4];
    __shared__ float rmax[64];
    __shared__ float rsum[64];

    int bid  = blockIdx.x;
    int qblk = bid & 3; int bph = bid >> 2;
    int h    = bph & 7; int bp  = bph >> 3;

    int tid = threadIdx.x, wave = tid >> 5, lane = tid & 31;
    int lm = lane & 15, kh = lane >> 4;

    const bf16* Q  = qb + ((size_t)bph * NSEQ + qblk * 64) * HD;
    const bf16* Kp = kb + (size_t)bph * NSEQ * HD;
    const bf16* Vt = vt + (size_t)bph * HD * NSEQ;

    // Phase 1: S = Qs * K^T   (64 x 256), wave -> mtile (w&3), col half (w>>2)
    {
        int mt = wave & 3, nh = wave >> 2;
        const bf16* aq = Q + (size_t)(mt * 16 + lm) * HD;
        v8f acc[8];
#pragma unroll
        for (int j = 0; j < 8; j++) acc[j] = zero8();
#pragma unroll
        for (int k = 0; k < HD; k += 32) {
            v16bf a = ld_frag(aq + k + kh * 8, aq + k + 16 + kh * 8);
#pragma unroll
            for (int j = 0; j < 8; j++) {
                const bf16* bk = Kp + (size_t)(nh * 128 + j * 16 + lm) * HD;
                v16bf b = ld_frag(bk + k + kh * 8, bk + k + 16 + kh * 8);
                acc[j] = wmma_bf16(a, b, acc[j]);
            }
        }
#pragma unroll
        for (int j = 0; j < 8; j++)
#pragma unroll
            for (int v = 0; v < 8; v++)
                Sb[mt * 16 + v + 8 * kh][nh * 128 + j * 16 + lm] = (bf16)acc[j][v];
    }
    __syncthreads();

    // Phase 2: row softmax (unnormalized exp; keep row sums)
    {
        int row = tid & 63, seg = tid >> 6;
        bf16* prow = &Sb[row][seg * 64];
        float mx = -1e30f;
        for (int i = 0; i < 64; i++) mx = fmaxf(mx, (float)prow[i]);
        red[row][seg] = mx;
        __syncthreads();
        if (tid < 64)
            rmax[tid] = fmaxf(fmaxf(red[tid][0], red[tid][1]),
                              fmaxf(red[tid][2], red[tid][3]));
        __syncthreads();
        float rm = rmax[row], sum = 0.f;
        for (int i = 0; i < 64; i++) {
            float e = __expf((float)prow[i] - rm);
            prow[i] = (bf16)e;
            sum += e;
        }
        red[row][seg] = sum;
        __syncthreads();
        if (tid < 64)
            rsum[tid] = red[tid][0] + red[tid][1] + red[tid][2] + red[tid][3];
        __syncthreads();
    }

    // Phase 3: O = P * V   (A-frags from LDS, B-frags from V^T)
    {
        int mt = wave >> 1, np = wave & 1;
        const bf16* arow = &Sb[mt * 16 + lm][0];
        v8f acc[2];
        acc[0] = zero8(); acc[1] = zero8();
#pragma unroll 2
        for (int k = 0; k < NSEQ; k += 32) {
            v16bf a = ld_frag(arow + k + kh * 8, arow + k + 16 + kh * 8);
#pragma unroll
            for (int j = 0; j < 2; j++) {
                const bf16* bv = Vt + (size_t)((np * 2 + j) * 16 + lm) * NSEQ;
                v16bf b = ld_frag(bv + k + kh * 8, bv + k + 16 + kh * 8);
                acc[j] = wmma_bf16(a, b, acc[j]);
            }
        }
#pragma unroll
        for (int v = 0; v < 8; v++) {
            int rl  = mt * 16 + v + 8 * kh;
            float inv = 1.0f / rsum[rl];
            int ng  = qblk * 64 + rl;
#pragma unroll
            for (int j = 0; j < 2; j++) {
                int e = (np * 2 + j) * 16 + lm;
                obuf[((size_t)bp * NSEQ + ng) * DIM + h * HD + e] =
                    (bf16)(acc[j][v] * inv);
            }
        }
    }
}

// ---------------- workspace layout (bytes), peak ~120 MB ----------------
static constexpr size_t OFF_WQKV  = 0;                 // 1536*512 bf16
static constexpr size_t OFF_WPROJ = 1572864;           // 512*512
static constexpr size_t OFF_W1    = 2097152;           // 2048*512
static constexpr size_t OFF_W2    = 4194304;           // 512*2048
static constexpr size_t OFF_H1    = 8388608;           // 16 MB: h1, reused as obuf
static constexpr size_t OFF_QKV   = 25165824;          // 48 MB: qkv, reused as x1 (32 MB f32)
static constexpr size_t OFF_X1    = 25165824;
static constexpr size_t OFF_QB    = 75497472;          // 16 MB, reused as h2
static constexpr size_t OFF_KB    = 92274688;          // 16 MB, reused as g-chunk
static constexpr size_t OFF_VT    = 109051904;         // 16 MB
static constexpr size_t OFF_H2    = 75497472;
static constexpr size_t OFF_G     = 92274688;

extern "C" void kernel_launch(void* const* d_in, const int* in_sizes, int n_in,
                              void* d_out, int out_size, void* d_ws, size_t ws_size,
                              hipStream_t stream) {
    const float* x      = (const float*)d_in[0];
    const float* ln1_g  = (const float*)d_in[1];
    const float* ln1_b  = (const float*)d_in[2];
    const float* w_qkv  = (const float*)d_in[3];
    const float* w_proj = (const float*)d_in[4];
    const float* b_proj = (const float*)d_in[5];
    const float* ln2_g  = (const float*)d_in[6];
    const float* ln2_b  = (const float*)d_in[7];
    const float* w1     = (const float*)d_in[8];
    const float* b1     = (const float*)d_in[9];
    const float* w2     = (const float*)d_in[10];
    const float* b2     = (const float*)d_in[11];
    float* out = (float*)d_out;

    char* ws = (char*)d_ws;
    bf16* WQKV = (bf16*)(ws + OFF_WQKV);
    bf16* WPRJ = (bf16*)(ws + OFF_WPROJ);
    bf16* W1B  = (bf16*)(ws + OFF_W1);
    bf16* W2B  = (bf16*)(ws + OFF_W2);
    bf16* h1   = (bf16*)(ws + OFF_H1);
    bf16* qkv  = (bf16*)(ws + OFF_QKV);
    float* x1  = (float*)(ws + OFF_X1);
    bf16* qb   = (bf16*)(ws + OFF_QB);
    bf16* kb   = (bf16*)(ws + OFF_KB);
    bf16* vt   = (bf16*)(ws + OFF_VT);
    bf16* obuf = h1;                       // reuse
    bf16* h2   = (bf16*)(ws + OFF_H2);     // reuse
    bf16* g    = (bf16*)(ws + OFF_G);      // reuse

    // weights -> bf16
    cvt_bf16<<<(3*DIM*DIM + 255)/256, 256, 0, stream>>>(w_qkv, WQKV, 3*DIM*DIM);
    cvt_bf16<<<(DIM*DIM   + 255)/256, 256, 0, stream>>>(w_proj, WPRJ, DIM*DIM);
    cvt_bf16<<<(HIDDEN*DIM+ 255)/256, 256, 0, stream>>>(w1, W1B, HIDDEN*DIM);
    cvt_bf16<<<(DIM*HIDDEN+ 255)/256, 256, 0, stream>>>(w2, W2B, DIM*HIDDEN);

    // LN1 -> h1 (bf16)
    ln_kernel<<<NTOK/8, 256, 0, stream>>>(x, ln1_g, ln1_b, h1);

    // QKV GEMM: (16384x512)*(1536x512)^T -> qkv bf16
    gemm_kernel<0><<<dim3(NTOK/128, (3*DIM)/256), 256, 0, stream>>>(
        h1, WQKV, 3*DIM, DIM, nullptr, nullptr, nullptr, qkv);

    // split + scale q + transpose v
    repack_qkv<<<(NTOK*DIM)/256, 256, 0, stream>>>(qkv, qb, kb, vt);

    // attention -> obuf (bf16, token-major (n, h*64+e))
    attn_kernel<<<BPH * (NSEQ/64), 256, 0, stream>>>(qb, kb, vt, obuf);

    // proj GEMM + bias + residual(x) -> x1 (f32)
    gemm_kernel<2><<<dim3(NTOK/128, DIM/256), 256, 0, stream>>>(
        obuf, WPRJ, DIM, DIM, b_proj, x, x1, nullptr);

    // LN2 -> h2 (bf16)
    ln_kernel<<<NTOK/8, 256, 0, stream>>>(x1, ln2_g, ln2_b, h2);

    // MLP, chunked over 4096-token slices to cap workspace
    for (int chunk = 0; chunk < 4; chunk++) {
        size_t toff = (size_t)chunk * 4096;
        gemm_kernel<1><<<dim3(4096/128, HIDDEN/256), 256, 0, stream>>>(
            h2 + toff * DIM, W1B, HIDDEN, DIM, b1, nullptr, nullptr, g);
        gemm_kernel<2><<<dim3(4096/128, DIM/256), 256, 0, stream>>>(
            g, W2B, DIM, HIDDEN, b2, x1 + toff * DIM, out + toff * DIM, nullptr);
    }
}
